// DepthNetv2_80642305950153
// MI455X (gfx1250) — matrix operand order repaired and checked by reading, not compiled, not run on hardware
//
#include <hip/hip_runtime.h>
#include <hip/hip_bf16.h>
#include <math.h>

// Problem constants (from setup_inputs): B=2, V=3, C=16, H=128, W=160, D=48
constexpr int B_ = 2;
constexpr int V_ = 3;
constexpr int C_ = 16;
constexpr int H_ = 128;
constexpr int W_ = 160;
constexpr int D_ = 48;
constexpr int HW_ = H_ * W_;
constexpr float EPS_BN = 1e-5f;

typedef __attribute__((ext_vector_type(16))) _Float16 v16h;
typedef __attribute__((ext_vector_type(8)))  float    v8f;

// ---------------------------------------------------------------------------
// Kernel 1: compose projection matrices and build per-(b,view) rot/trans.
// proj = compose(src) @ inv(compose(ref));  rot = proj[:3,:3], trans = proj[:3,3]
// ---------------------------------------------------------------------------
__device__ inline void compose4(const float* p /* [2,4,4] */, float* out /* [4,4] */) {
  const float* E = p;        // extrinsic-like
  const float* K = p + 16;   // intrinsic-like
  for (int i = 0; i < 16; ++i) out[i] = E[i];
  float tmp[12];
  for (int i = 0; i < 3; ++i)
    for (int j = 0; j < 4; ++j) {
      float s = 0.f;
      for (int k = 0; k < 3; ++k) s += K[i * 4 + k] * E[k * 4 + j];
      tmp[i * 4 + j] = s;
    }
  for (int i = 0; i < 12; ++i) out[i] = tmp[i];
}

__device__ inline void matmul4(const float* A, const float* Bm, float* Cm) {
  for (int i = 0; i < 4; ++i)
    for (int j = 0; j < 4; ++j) {
      float s = 0.f;
      for (int k = 0; k < 4; ++k) s += A[i * 4 + k] * Bm[k * 4 + j];
      Cm[i * 4 + j] = s;
    }
}

__device__ inline void inv4(const float* M, float* out) {
  float a[4][8];
  for (int i = 0; i < 4; ++i)
    for (int j = 0; j < 4; ++j) { a[i][j] = M[i * 4 + j]; a[i][j + 4] = (i == j) ? 1.f : 0.f; }
  for (int c = 0; c < 4; ++c) {
    int p = c; float best = fabsf(a[c][c]);
    for (int r = c + 1; r < 4; ++r) { float v = fabsf(a[r][c]); if (v > best) { best = v; p = r; } }
    if (p != c) for (int j = 0; j < 8; ++j) { float t = a[c][j]; a[c][j] = a[p][j]; a[p][j] = t; }
    float ip = 1.f / a[c][c];
    for (int j = 0; j < 8; ++j) a[c][j] *= ip;
    for (int r = 0; r < 4; ++r) {
      if (r == c) continue;
      float f = a[r][c];
      for (int j = 0; j < 8; ++j) a[r][j] -= f * a[c][j];
    }
  }
  for (int i = 0; i < 4; ++i)
    for (int j = 0; j < 4; ++j) out[i * 4 + j] = a[i][j + 4];
}

__global__ void setup_proj_kernel(const float* __restrict__ projm /* [B,V,2,4,4] */,
                                  float* __restrict__ rt /* [B,V-1,12] */) {
  if (threadIdx.x != 0 || blockIdx.x != 0) return;
  for (int b = 0; b < B_; ++b) {
    float refc[16], refinv[16];
    compose4(projm + ((b * V_ + 0) * 2) * 16, refc);
    inv4(refc, refinv);
    for (int vi = 0; vi < V_ - 1; ++vi) {
      float srcc[16], proj[16];
      compose4(projm + ((b * V_ + vi + 1) * 2) * 16, srcc);
      matmul4(srcc, refinv, proj);
      float* o = rt + (b * (V_ - 1) + vi) * 12;
      o[0] = proj[0];  o[1] = proj[1];  o[2]  = proj[2];
      o[3] = proj[4];  o[4] = proj[5];  o[5]  = proj[6];
      o[6] = proj[8];  o[7] = proj[9];  o[8]  = proj[10];
      o[9] = proj[3];  o[10] = proj[7]; o[11] = proj[11];
    }
  }
}

// ---------------------------------------------------------------------------
// Kernel 2: homography warp + bilinear sample + channel-mean similarity.
// One thread per (view,b,d,hw). Writes sim volume [(V-1)*B, D, HW].
// ---------------------------------------------------------------------------
__global__ void warp_sim_kernel(const float* __restrict__ feat   /* [B,V,C,H,W] */,
                                const float* __restrict__ depthv /* [B,D,H,W]   */,
                                const float* __restrict__ rt     /* [B,V-1,12]  */,
                                float* __restrict__ simv /* [(V-1)*B, D, HW] */) {
  int hw = blockIdx.x * blockDim.x + threadIdx.x;
  if (hw >= HW_) return;
  int d  = blockIdx.y;
  int bz = blockIdx.z;
  int b  = bz % B_;
  int vi = bz / B_;

  const float* RT = rt + (b * (V_ - 1) + vi) * 12;
  float xf = (float)(hw % W_);
  float yf = (float)(hw / W_);
  float dep = depthv[(b * D_ + d) * HW_ + hw];

  float px = RT[0] * xf + RT[1] * yf + RT[2];
  float py = RT[3] * xf + RT[4] * yf + RT[5];
  float pz = RT[6] * xf + RT[7] * yf + RT[8];
  px = px * dep + RT[9];
  py = py * dep + RT[10];
  pz = pz * dep + RT[11];
  float ix = px / pz;
  float iy = py / pz;

  float x0 = floorf(ix), y0 = floorf(iy);
  float wgt[4]; int idx[4]; int t = 0;
  #pragma unroll
  for (int dyi = 0; dyi < 2; ++dyi) {
    #pragma unroll
    for (int dxi = 0; dxi < 2; ++dxi) {
      float xi = x0 + (float)dxi;
      float yi = y0 + (float)dyi;
      float wv = (1.f - fabsf(ix - xi)) * (1.f - fabsf(iy - yi));
      bool valid = (xi >= 0.f) && (xi <= (float)(W_ - 1)) &&
                   (yi >= 0.f) && (yi <= (float)(H_ - 1));
      float xc = fminf(fmaxf(xi, 0.f), (float)(W_ - 1));
      float yc = fminf(fmaxf(yi, 0.f), (float)(H_ - 1));
      idx[t] = (int)yc * W_ + (int)xc;
      wgt[t] = valid ? wv : 0.f;
      ++t;
    }
  }

  const float* F = feat + (size_t)((b * V_ + (vi + 1)) * C_) * HW_;
  const float* R = feat + (size_t)((b * V_ + 0) * C_) * HW_;
  float sim = 0.f;
  #pragma unroll
  for (int c = 0; c < C_; ++c) {
    const float* Fc = F + c * HW_;
    float wval = wgt[0] * Fc[idx[0]] + wgt[1] * Fc[idx[1]] +
                 wgt[2] * Fc[idx[2]] + wgt[3] * Fc[idx[3]];
    sim += wval * R[c * HW_ + hw];
  }
  sim *= (1.f / (float)C_);
  simv[((size_t)(vi * B_ + b) * D_ + d) * HW_ + hw] = sim;
}

// ---------------------------------------------------------------------------
// Kernel 3: pixelwise MLP (1->16->8->1, BN+ReLU, sigmoid) + max over D.
// The 8x16 middle layer is computed with V_WMMA_F32_16X16X32_F16:
//   A = w1 zero-padded 16x32 (rows 8..15 and K>=16 zero)
//   B = h0 activations: K rows 0..15 = 16 channels, cols = 16 pixels
//   D rows 0..7 (lanes 0..15, VGPRs 0..7) = h1[o][pixel]
// One wave handles 16 pixels; lanes 16..31 carry the zero K=16..31 rows.
// Depth loop is unrolled x2 -> two independent WMMA chains per iteration to
// cover the f16-WMMA RAW hazard window. sigmoid is hoisted out of the loop
// (monotonic: max_d sigmoid(y) == sigmoid(max_d y)).
// Grid is exact (no early exit) so EXEC is all-1s at every WMMA.
// ---------------------------------------------------------------------------
__global__ void mlp_vw_kernel(const float* __restrict__ simv /* [(V-1)*B, D, HW] */,
                              const float* __restrict__ w0, const float* __restrict__ g0,
                              const float* __restrict__ bb0, const float* __restrict__ m0,
                              const float* __restrict__ vv0,
                              const float* __restrict__ w1, const float* __restrict__ g1,
                              const float* __restrict__ bb1, const float* __restrict__ m1,
                              const float* __restrict__ vv1,
                              const float* __restrict__ w2, const float* __restrict__ b2,
                              float* __restrict__ vw_out /* [B, V-1, HW] */) {
  const int lane   = threadIdx.x & 31;
  const int wave   = threadIdx.x >> 5;
  const int laneM  = lane & 15;
  const int laneHi = lane >> 4;          // 0: K=0..15 (A and B), 1: padding rows
  const int b  = blockIdx.y;
  const int vi = blockIdx.z;
  const int pix = blockIdx.x * ((int)blockDim.x / 2) + wave * 16 + laneM; // exact cover of HW_

  // Folded BN coefficients (uniform across lanes).
  float a0[16], t0[16];
  #pragma unroll
  for (int c = 0; c < 16; ++c) {
    float s = g0[c] * rsqrtf(vv0[c] + EPS_BN);
    a0[c] = w0[c] * s;                  // w0 is [16,1]
    t0[c] = bb0[c] - m0[c] * s;
  }
  float s1[8], t1[8], w2r[8];
  #pragma unroll
  for (int o = 0; o < 8; ++o) {
    float s = g1[o] * rsqrtf(vv1[o] + EPS_BN);
    s1[o] = s;
    t1[o] = bb1[o] - m1[o] * s;
    w2r[o] = w2[o];
  }
  const float bias2 = b2[0];

  // A matrix: contiguous K order, lane (M=laneM), half h -> K = laneHi*16+h (laneHi=1 is K padding).
  v16h A;
  #pragma unroll
  for (int h = 0; h < 16; ++h) A[h] = (_Float16)0.f;
  if (laneM < 8 && laneHi == 0) {
    #pragma unroll
    for (int h = 0; h < 16; ++h) A[h] = (_Float16)w1[laneM * 16 + h];
  }

  const float keep = (laneHi == 0) ? 1.f : 0.f;  // zero the K=16..31 B rows without divergence
  const float* simp = simv + (size_t)(vi * B_ + b) * D_ * HW_;

  float ymax = -3.4e38f;
  for (int d = 0; d < D_; d += 2) {
    float x0v = simp[(size_t)d * HW_ + pix];           // hi lanes load a duplicate (harmless)
    float x1v = simp[(size_t)(d + 1) * HW_ + pix];
    v16h Bm0, Bm1;
    #pragma unroll
    for (int c = 0; c < 16; ++c) {
      float h0a = a0[c] * x0v + t0[c];
      float h0b = a0[c] * x1v + t0[c];
      h0a = h0a > 0.f ? h0a : 0.f;
      h0b = h0b > 0.f ? h0b : 0.f;
      Bm0[c] = (_Float16)(h0a * keep);                 // B[K=c][N=laneM]
      Bm1[c] = (_Float16)(h0b * keep);
    }
    v8f C0 = {};
    v8f C1 = {};
    C0 = __builtin_amdgcn_wmma_f32_16x16x32_f16(false, A, false, Bm0,
                                                (short)0, C0, false, false);
    C1 = __builtin_amdgcn_wmma_f32_16x16x32_f16(false, A, false, Bm1,
                                                (short)0, C1, false, false);
    // Lane p (0..15): C[o] = h1_pre[o][p].
    float ya = bias2, yb = bias2;
    #pragma unroll
    for (int o = 0; o < 8; ++o) {
      float h1a = s1[o] * C0[o] + t1[o];
      float h1b = s1[o] * C1[o] + t1[o];
      h1a = h1a > 0.f ? h1a : 0.f;
      h1b = h1b > 0.f ? h1b : 0.f;
      ya += w2r[o] * h1a;
      yb += w2r[o] * h1b;
    }
    ymax = fmaxf(ymax, fmaxf(ya, yb));
  }
  // sigmoid is monotonic: apply once to the max logit.
  float vw = 1.f / (1.f + expf(-ymax));
  if (laneHi == 0)
    vw_out[((size_t)b * (V_ - 1) + vi) * HW_ + pix] = vw;
}

// ---------------------------------------------------------------------------
// Kernel 4: similarity = sum_v(sim_v * vw_v) / (1e-5 + sum_v vw_v)
// ---------------------------------------------------------------------------
__global__ void combine_kernel(const float* __restrict__ simv /* [(V-1)*B,D,HW] */,
                               const float* __restrict__ vw   /* [B,V-1,HW]      */,
                               float* __restrict__ simil      /* [B,D,HW]        */) {
  int idx = blockIdx.x * blockDim.x + threadIdx.x;
  const int total = B_ * D_ * HW_;
  if (idx >= total) return;
  int b   = idx / (D_ * HW_);
  int rem = idx % (D_ * HW_);
  int hw  = rem % HW_;
  float num = 0.f, den = 1e-5f;
  #pragma unroll
  for (int vi = 0; vi < V_ - 1; ++vi) {
    float s  = simv[(size_t)(vi * B_ + b) * (D_ * HW_) + rem];
    float wv = vw[((size_t)b * (V_ - 1) + vi) * HW_ + hw];
    num += s * wv;
    den += wv;
  }
  simil[idx] = num / den;
}

// ---------------------------------------------------------------------------
// Kernel 5: 3x3x3 SAME conv (1ch), softmax over D, argmax -> depth/conf/prob.
// One thread per (b,hw).
// ---------------------------------------------------------------------------
__global__ void conv_softmax_kernel(const float* __restrict__ simil  /* [B,D,HW] */,
                                    const float* __restrict__ depthv /* [B,D,HW] */,
                                    const float* __restrict__ regw   /* 27 */,
                                    const float* __restrict__ regb   /* 1  */,
                                    float* __restrict__ out_depth,
                                    float* __restrict__ out_conf,
                                    float* __restrict__ out_prob) {
  int hw = blockIdx.x * blockDim.x + threadIdx.x;
  if (hw >= HW_) return;
  int b = blockIdx.y;
  int h = hw / W_, w = hw % W_;

  float kw[27];
  #pragma unroll
  for (int i = 0; i < 27; ++i) kw[i] = regw[i];
  float kb = regb[0];

  const float* S = simil + (size_t)b * D_ * HW_;
  float cost[D_];
  float cmax = -3.4e38f;
  for (int d = 0; d < D_; ++d) {
    float acc = kb;
    #pragma unroll
    for (int kd = 0; kd < 3; ++kd) {
      int zz = d + kd - 1;
      if (zz < 0 || zz >= D_) continue;
      #pragma unroll
      for (int kh = 0; kh < 3; ++kh) {
        int yy = h + kh - 1;
        if (yy < 0 || yy >= H_) continue;
        #pragma unroll
        for (int kx = 0; kx < 3; ++kx) {
          int xx = w + kx - 1;
          if (xx < 0 || xx >= W_) continue;
          acc += kw[(kd * 3 + kh) * 3 + kx] * S[(size_t)zz * HW_ + yy * W_ + xx];
        }
      }
    }
    cost[d] = acc;
    cmax = fmaxf(cmax, acc);
  }
  float sum = 0.f;
  for (int d = 0; d < D_; ++d) { float e = expf(cost[d] - cmax); cost[d] = e; sum += e; }
  float inv = 1.f / sum;
  int best = 0; float bp = -1.f;
  for (int d = 0; d < D_; ++d) {
    float p = cost[d] * inv;
    out_prob[((size_t)b * D_ + d) * HW_ + hw] = p;
    if (p > bp) { bp = p; best = d; }   // first-max wins, matches jnp.argmax
  }
  out_depth[(size_t)b * HW_ + hw] = depthv[((size_t)b * D_ + best) * HW_ + hw];
  out_conf[(size_t)b * HW_ + hw]  = bp;
}

// ---------------------------------------------------------------------------
// Launch
// ---------------------------------------------------------------------------
extern "C" void kernel_launch(void* const* d_in, const int* in_sizes, int n_in,
                              void* d_out, int out_size, void* d_ws, size_t ws_size,
                              hipStream_t stream) {
  const float* features = (const float*)d_in[0];   // [B,V,C,H,W]
  const float* projm    = (const float*)d_in[1];   // [B,V,2,4,4]
  const float* depthv   = (const float*)d_in[2];   // [B,D,H,W]
  const float* w0 = (const float*)d_in[3];
  const float* g0 = (const float*)d_in[4];
  const float* b0 = (const float*)d_in[5];
  const float* m0 = (const float*)d_in[6];
  const float* v0 = (const float*)d_in[7];
  const float* w1 = (const float*)d_in[8];
  const float* g1 = (const float*)d_in[9];
  const float* b1 = (const float*)d_in[10];
  const float* m1 = (const float*)d_in[11];
  const float* v1 = (const float*)d_in[12];
  const float* w2 = (const float*)d_in[13];
  const float* b2 = (const float*)d_in[14];
  const float* regw = (const float*)d_in[15];
  const float* regb = (const float*)d_in[16];
  (void)in_sizes; (void)n_in; (void)out_size; (void)ws_size;

  float* out = (float*)d_out;
  float* out_depth = out;                                // [B,H,W]
  float* out_conf  = out + (size_t)B_ * HW_;             // [B,H,W]
  float* out_prob  = out + (size_t)2 * B_ * HW_;         // [B,D,H,W]
  float* out_vw    = out + (size_t)2 * B_ * HW_ + (size_t)B_ * D_ * HW_; // [B,V-1,H,W]

  float* wsf   = (float*)d_ws;
  float* rt    = wsf;                                    // 64 floats (48 used)
  float* simv  = wsf + 64;                               // (V-1)*B*D*HW
  float* simil = simv + (size_t)(V_ - 1) * B_ * D_ * HW_; // B*D*HW

  // 1) projection composition + inverse (tiny)
  setup_proj_kernel<<<dim3(1), dim3(1), 0, stream>>>(projm, rt);

  // 2) warp + similarity volumes
  dim3 g2((HW_ + 255) / 256, D_, B_ * (V_ - 1));
  warp_sim_kernel<<<g2, dim3(256), 0, stream>>>(features, depthv, rt, simv);

  // 3) pixelwise MLP with WMMA -> view weights (written into d_out slice)
  dim3 g3(HW_ / 128, B_, V_ - 1);  // 256 threads = 8 waves * 16 pixels = 128 px/block, exact
  mlp_vw_kernel<<<g3, dim3(256), 0, stream>>>(simv, w0, g0, b0, m0, v0,
                                              w1, g1, b1, m1, v1, w2, b2, out_vw);

  // 4) weighted fusion of similarity volumes
  const int n4 = B_ * D_ * HW_;
  combine_kernel<<<dim3((n4 + 255) / 256), dim3(256), 0, stream>>>(simv, out_vw, simil);

  // 5) 3D conv + softmax + argmax -> depth / confidence / prob volume
  dim3 g5((HW_ + 255) / 256, B_);
  conv_softmax_kernel<<<g5, dim3(256), 0, stream>>>(simil, depthv, regw, regb,
                                                    out_depth, out_conf, out_prob);
}
